// MaskedAttention_61512521614019
// MI455X (gfx1250) — compile-verified, matrix operand order checked by reading, not run on hardware
//
#include <hip/hip_runtime.h>

typedef __attribute__((ext_vector_type(16))) __bf16 v16bf;
typedef __attribute__((ext_vector_type(8)))  __bf16 v8bf;
typedef __attribute__((ext_vector_type(4)))  __bf16 v4bf;
typedef __attribute__((ext_vector_type(8)))  float  v8f;
typedef __attribute__((ext_vector_type(4)))  int    v4i;

#define EMBED 1024
#define SEQ   2048
#define NB    2
#define NH    16
#define DH    64

static __device__ __forceinline__ v8f wmma_bf16(v16bf a, v16bf b, v8f c) {
  // D = A(16x32 bf16) x B(32x16 bf16) + C(16x16 f32)
  return __builtin_amdgcn_wmma_f32_16x16x32_bf16(false, a, false, b, (short)0, c,
                                                 false, false);
}

// Load one 16x32 (bf16) A/B operand fragment from row-major global memory.
// Row for this lane = (lane & 15); K swizzle per CDNA5 16-bit layout:
//   lanes 0-15 hold K = {0..7, 16..23}, lanes 16-31 hold K = {8..15, 24..31}.
static __device__ __forceinline__ v16bf load_frag(const __bf16* rowptr, int k0,
                                                  int lane) {
  const int off = (lane & 16) ? 8 : 0;
  union { v16bf v; v8bf h[2]; } u;
  u.h[0] = *(const v8bf*)(rowptr + k0 + off);
  u.h[1] = *(const v8bf*)(rowptr + k0 + off + 16);
  return u.v;
}

// ---------------------------------------------------------- async copy to LDS
// GLOBAL_LOAD_ASYNC_TO_LDS_B128: per-lane 16B global->LDS, tracked by ASYNCcnt.
static __device__ __forceinline__ void async_ld_b128(const __bf16* g, __bf16* l) {
#if __has_builtin(__builtin_amdgcn_global_load_async_to_lds_b128)
  __builtin_amdgcn_global_load_async_to_lds_b128(
      (__attribute__((address_space(1))) v4i*)(__bf16*)g,
      (__attribute__((address_space(3))) v4i*)l, 0, 0);
#else
  // VDST = raw LDS byte offset (addrspace(3) ptrtoint), VADDR = 64-bit global.
  const unsigned lds_off =
      (unsigned)(unsigned long long)(__attribute__((address_space(3))) __bf16*)l;
  asm volatile("global_load_async_to_lds_b128 %0, %1, off" ::"v"(lds_off),
               "v"((unsigned long long)(size_t)g)
               : "memory");
#endif
}

static __device__ __forceinline__ void wait_async0() {
#if __has_builtin(__builtin_amdgcn_s_wait_asynccnt)
  __builtin_amdgcn_s_wait_asynccnt(0);
#else
  asm volatile("s_wait_asynccnt 0x0" ::: "memory");
#endif
}

// ---------------------------------------------------------------- fp32 -> bf16
__global__ __launch_bounds__(256) void cvt_f32_bf16(const float* __restrict__ in,
                                                    __bf16* __restrict__ out,
                                                    int n4) {
  const int i = blockIdx.x * 256 + threadIdx.x;
  if (i < n4) {
    const float4 v = ((const float4*)in)[i];
    v4bf o;
    o.x = (__bf16)v.x; o.y = (__bf16)v.y; o.z = (__bf16)v.z; o.w = (__bf16)v.w;
    ((v4bf*)out)[i] = o;
  }
}

// --------------------------------------------------- C = A @ W^T + bias (WMMA)
// A: [M,1024] bf16 row-major. W: [1024,1024] bf16 row-major [n][k] (B[k,n] =
// W[n,k]). Block = 128 threads (4 waves), tile 128M x 64N x 32K, double-
// buffered LDS filled with async-to-LDS copies; wave computes 32M x 64N
// (8 WMMAs per K-step from 12 ds_load_b128).
enum { OUT_BF16 = 0, OUT_BF16_VT = 1, OUT_F32 = 2 };

#define BM 128
#define BN 64
#define BK 32

template <int MODE>
__global__ __launch_bounds__(128) void gemm_xwt_bias(
    const __bf16* __restrict__ A, const __bf16* __restrict__ W,
    const float* __restrict__ bias, void* __restrict__ out) {
  __shared__ __align__(16) __bf16 sA[2][BM][BK];  // 2 x 8 KB
  __shared__ __align__(16) __bf16 sW[2][BN][BK];  // 2 x 4 KB

  const int tid  = threadIdx.x;
  const int lane = tid & 31;
  const int wave = tid >> 5;  // 0..3
  const int r    = lane & 15;
  const int mh   = (lane & 16) ? 8 : 0;
  const int mblock = blockIdx.y * BM;
  const int nbase  = blockIdx.x * BN;

  // Staging map: one chunk = 16 B = 8 bf16. A tile = 512 chunks, W tile = 256;
  // 128 threads -> 4 A-chunks + 2 W-chunks per thread per K-step.
  auto stage = [&](int k0, int buf) {
#pragma unroll
    for (int cc = 0; cc < 4; ++cc) {
      const int c = tid + cc * 128;
      const int row = c >> 2, col = (c & 3) * 8;
      async_ld_b128(A + (size_t)(mblock + row) * EMBED + k0 + col,
                    &sA[buf][row][col]);
    }
#pragma unroll
    for (int cc = 0; cc < 2; ++cc) {
      const int c = tid + cc * 128;
      const int row = c >> 2, col = (c & 3) * 8;
      async_ld_b128(W + (size_t)(nbase + row) * EMBED + k0 + col,
                    &sW[buf][row][col]);
    }
  };

  v8f acc[8] = {};  // [mt*4 + j], mt in {0,1}, j in 0..3
  stage(0, 0);

  const int off = (lane & 16) ? 8 : 0;
  for (int kt = 0; kt < EMBED / BK; ++kt) {
    wait_async0();     // this wave's async writes to LDS are done
    __syncthreads();   // everyone's writes visible; prior reads retired
    if (kt + 1 < EMBED / BK) stage((kt + 1) * BK, (kt + 1) & 1);

    const int buf = kt & 1;
    union { v16bf v; v8bf h[2]; } ua0, ua1, ub;
    ua0.h[0] = *(const v8bf*)&sA[buf][wave * 32 + r][off];
    ua0.h[1] = *(const v8bf*)&sA[buf][wave * 32 + r][off + 16];
    ua1.h[0] = *(const v8bf*)&sA[buf][wave * 32 + 16 + r][off];
    ua1.h[1] = *(const v8bf*)&sA[buf][wave * 32 + 16 + r][off + 16];
#pragma unroll
    for (int j = 0; j < 4; ++j) {
      ub.h[0] = *(const v8bf*)&sW[buf][j * 16 + r][off];
      ub.h[1] = *(const v8bf*)&sW[buf][j * 16 + r][off + 16];
      acc[j]     = wmma_bf16(ua0.v, ub.v, acc[j]);
      acc[4 + j] = wmma_bf16(ua1.v, ub.v, acc[4 + j]);
    }
  }

#pragma unroll
  for (int mt = 0; mt < 2; ++mt)
#pragma unroll
    for (int j = 0; j < 4; ++j) {
      const int n = nbase + j * 16 + r;
      const float bv = bias[n];
#pragma unroll
      for (int i = 0; i < 8; ++i) {
        const int m = mblock + wave * 32 + mt * 16 + mh + i;
        const float val = acc[mt * 4 + j][i] + bv;
        if (MODE == OUT_F32) {
          ((float*)out)[(size_t)m * EMBED + n] = val;
        } else if (MODE == OUT_BF16) {
          ((__bf16*)out)[(size_t)m * EMBED + n] = (__bf16)val;
        } else {  // V^T layout: [B, H, DH, SEQ]
          const int b = m >> 11, s = m & (SEQ - 1);
          const int h = n >> 6, d = n & (DH - 1);
          ((__bf16*)out)[((size_t)(b * NH + h) * DH + d) * SEQ + s] = (__bf16)val;
        }
      }
    }
}

// ------------------------------------------------- causal flash attention core
// Q,K: [B*S, 1024] bf16 (heads interleaved).  Vt: [B,H,DH,S] bf16.
// Wave = one 16-row Q tile of one (b,h); block = 4 waves = 64 q rows.
__global__ __launch_bounds__(128) void attn_causal(
    const __bf16* __restrict__ Q, const __bf16* __restrict__ Km,
    const __bf16* __restrict__ Vt, __bf16* __restrict__ O) {
  __shared__ __align__(16) __bf16 pbuf[4][16][32];

  const int lane = threadIdx.x & 31;
  const int wave = threadIdx.x >> 5;
  const int r    = lane & 15;
  const int mh   = (lane & 16) ? 8 : 0;
  const int bh   = blockIdx.x;
  const int b = bh >> 4, h = bh & 15;
  const int qbase = blockIdx.y * 64 + wave * 16;

  const __bf16* qrow  = Q + ((size_t)(b * SEQ + qbase + r) * EMBED + h * DH);
  const __bf16* khead = Km + ((size_t)(b * SEQ) * EMBED + h * DH);
  const __bf16* vhead = Vt + (size_t)(b * NH + h) * DH * SEQ;

  const v16bf a0 = load_frag(qrow, 0, lane);   // Q d = 0..31
  const v16bf a1 = load_frag(qrow, 32, lane);  // Q d = 32..63

  v8f acc0 = {}, acc1 = {}, acc2 = {}, acc3 = {};
  float mrun[8], lrun[8];
#pragma unroll
  for (int i = 0; i < 8; ++i) { mrun[i] = -__builtin_inff(); lrun[i] = 0.f; }

  const int qend = qbase + 15;
  for (int k0 = 0; k0 <= qend; k0 += 32) {
    const int kr0 = k0 + r;  // <= SEQ-1 always
    int kr1 = k0 + 16 + r;
    if (kr1 > SEQ - 1) kr1 = SEQ - 1;  // masked columns anyway
    const __bf16* krow0 = khead + (size_t)kr0 * EMBED;
    const __bf16* krow1 = khead + (size_t)kr1 * EMBED;

    v8f s0 = {}, s1 = {};
    s0 = wmma_bf16(a0, load_frag(krow0, 0, lane), s0);
    s0 = wmma_bf16(a1, load_frag(krow0, 32, lane), s0);
    s1 = wmma_bf16(a0, load_frag(krow1, 0, lane), s1);
    s1 = wmma_bf16(a1, load_frag(krow1, 32, lane), s1);

    // Online softmax; rows m=i (lanes 0-15) / m=i+8 (lanes 16-31), n across
    // the 16-lane halves, so xor-reductions with masks 1..8 stay in-half.
#pragma unroll
    for (int i = 0; i < 8; ++i) {
      const int m = qbase + mh + i;
      float x0 = s0[i] * 0.125f;  // 1/sqrt(64)
      float x1 = s1[i] * 0.125f;
      if (k0 + r > m) x0 = -__builtin_inff();
      if (k0 + 16 + r > m) x1 = -__builtin_inff();
      float tm = fmaxf(x0, x1);
#pragma unroll
      for (int d = 1; d < 16; d <<= 1) tm = fmaxf(tm, __shfl_xor(tm, d, 32));
      const float mnew  = fmaxf(mrun[i], tm);
      const float alpha = __expf(mrun[i] - mnew);
      const float p0 = __expf(x0 - mnew);
      const float p1 = __expf(x1 - mnew);
      float rs = p0 + p1;
#pragma unroll
      for (int d = 1; d < 16; d <<= 1) rs += __shfl_xor(rs, d, 32);
      lrun[i] = lrun[i] * alpha + rs;
      mrun[i] = mnew;
      acc0[i] *= alpha; acc1[i] *= alpha; acc2[i] *= alpha; acc3[i] *= alpha;
      pbuf[wave][mh + i][r]      = (__bf16)p0;
      pbuf[wave][mh + i][16 + r] = (__bf16)p1;
    }

    // Re-read P (16x32) from LDS in A-fragment layout (intra-wave dependency:
    // compiler inserts s_wait_dscnt; no barrier needed).
    v16bf pa;
    {
      const int off = (lane & 16) ? 8 : 0;
      union { v16bf v; v8bf hh[2]; } u;
      u.hh[0] = *(const v8bf*)&pbuf[wave][r][off];
      u.hh[1] = *(const v8bf*)&pbuf[wave][r][off + 16];
      pa = u.v;
    }

    // V^T B-fragments: lane row = output d, K along s (contiguous).
    const int voff = (lane & 16) ? 8 : 0;
    int sa = k0 + voff;      if (sa > SEQ - 8) sa = SEQ - 8;
    int sb = k0 + voff + 16; if (sb > SEQ - 8) sb = SEQ - 8;
#pragma unroll
    for (int j = 0; j < 4; ++j) {
      const __bf16* vrow = vhead + (size_t)(j * 16 + r) * SEQ;
      union { v16bf v; v8bf hh[2]; } u;
      u.hh[0] = *(const v8bf*)(vrow + sa);
      u.hh[1] = *(const v8bf*)(vrow + sb);
      v8f& acc = (j == 0) ? acc0 : (j == 1) ? acc1 : (j == 2) ? acc2 : acc3;
      acc = wmma_bf16(pa, u.v, acc);
    }
  }

  __bf16* orow = O + ((size_t)(b * SEQ) * EMBED + h * DH);
  v8f oc[4] = {acc0, acc1, acc2, acc3};
#pragma unroll
  for (int j = 0; j < 4; ++j)
#pragma unroll
    for (int i = 0; i < 8; ++i) {
      const float o = oc[j][i] / lrun[i];
      orow[(size_t)(qbase + mh + i) * EMBED + j * 16 + r] = (__bf16)o;
    }
}

// ------------------------------------------------------------------- launcher
extern "C" void kernel_launch(void* const* d_in, const int* in_sizes, int n_in,
                              void* d_out, int out_size, void* d_ws,
                              size_t ws_size, hipStream_t stream) {
  (void)in_sizes; (void)n_in; (void)out_size; (void)ws_size;
  const float* x  = (const float*)d_in[0];
  const float* Wq = (const float*)d_in[1];
  const float* bq = (const float*)d_in[2];
  const float* Wk = (const float*)d_in[3];
  const float* bk = (const float*)d_in[4];
  const float* Wv = (const float*)d_in[5];
  const float* bv = (const float*)d_in[6];
  const float* Wo = (const float*)d_in[7];
  const float* bo = (const float*)d_in[8];

  char* ws = (char*)d_ws;
  const size_t MB = 1024u * 1024u;
  const size_t M  = (size_t)NB * SEQ;       // 4096 rows
  const size_t xN = M * EMBED;              // 4M elems (8 MB bf16)
  const size_t wN = (size_t)EMBED * EMBED;  // 1M elems (2 MB bf16)

  __bf16* xbf = (__bf16*)(ws + 0 * MB);
  __bf16* wqb = (__bf16*)(ws + 8 * MB);
  __bf16* wkb = (__bf16*)(ws + 10 * MB);
  __bf16* wvb = (__bf16*)(ws + 12 * MB);
  __bf16* wob = (__bf16*)(ws + 14 * MB);
  __bf16* Qb  = (__bf16*)(ws + 16 * MB);
  __bf16* Kb  = (__bf16*)(ws + 24 * MB);
  __bf16* Vtb = (__bf16*)(ws + 32 * MB);
  __bf16* Ob  = (__bf16*)(ws + 40 * MB);  // 48 MB total

  cvt_f32_bf16<<<(unsigned)((xN / 4 + 255) / 256), 256, 0, stream>>>(x, xbf, (int)(xN / 4));
  cvt_f32_bf16<<<(unsigned)((wN / 4 + 255) / 256), 256, 0, stream>>>(Wq, wqb, (int)(wN / 4));
  cvt_f32_bf16<<<(unsigned)((wN / 4 + 255) / 256), 256, 0, stream>>>(Wk, wkb, (int)(wN / 4));
  cvt_f32_bf16<<<(unsigned)((wN / 4 + 255) / 256), 256, 0, stream>>>(Wv, wvb, (int)(wN / 4));
  cvt_f32_bf16<<<(unsigned)((wN / 4 + 255) / 256), 256, 0, stream>>>(Wo, wob, (int)(wN / 4));

  const dim3 gg(EMBED / BN, (unsigned)(M / BM));  // (16, 32)
  gemm_xwt_bias<OUT_BF16><<<gg, 128, 0, stream>>>(xbf, wqb, bq, Qb);
  gemm_xwt_bias<OUT_BF16><<<gg, 128, 0, stream>>>(xbf, wkb, bk, Kb);
  gemm_xwt_bias<OUT_BF16_VT><<<gg, 128, 0, stream>>>(xbf, wvb, bv, Vtb);

  attn_causal<<<dim3(NB * NH, SEQ / 64), 128, 0, stream>>>(Qb, Kb, Vtb, Ob);

  gemm_xwt_bias<OUT_F32><<<gg, 128, 0, stream>>>(Ob, wob, bo, d_out);
}
